// BigramLM_47493748359286
// MI455X (gfx1250) — matrix operand order repaired and checked by reading, not compiled, not run on hardware
//
#include <hip/hip_runtime.h>

typedef _Float16 v8h  __attribute__((ext_vector_type(8)));
typedef _Float16 v16h __attribute__((ext_vector_type(16)));
typedef float    v8f  __attribute__((ext_vector_type(8)));

#define GPT_L  12
#define GPT_D  768
#define GPT_H  12
#define GPT_HD 64
#define GPT_V  50257
#define GPT_VP 50304               // GPT_V padded to multiple of 64
#define GPT_B  4
#define GPT_T  1024
#define GPT_BT (GPT_B * GPT_T)     // 4096
#define GPT_DF (4 * GPT_D)         // 3072

__device__ __forceinline__ v8h ld8(const _Float16* p) {
  return *reinterpret_cast<const v8h*>(p);
}
__device__ __forceinline__ v16h cat16(v8h a, v8h b) {
  return __builtin_shufflevector(a, b, 0,1,2,3,4,5,6,7,8,9,10,11,12,13,14,15);
}
__device__ __forceinline__ v8f wmma_f16(v16h a, v16h b, v8f c) {
  return __builtin_amdgcn_wmma_f32_16x16x32_f16(false, a, false, b, (short)0, c, false, false);
}

// ---------------------------------------------------------------------------
// Weight prep: f32 row-major W[K][N] -> f16 column-major Wt[Npad][K], zero
// padded in N. Tiled 32x32 LDS transpose, coalesced on both sides.
// Grid: (Npad/32, K/32), block (32,8).
// ---------------------------------------------------------------------------
__global__ __launch_bounds__(256)
void wtrans_kernel(const float* __restrict__ in, _Float16* __restrict__ out,
                   int K, int N)
{
  __shared__ float tile[32][33];
  const int nb = blockIdx.x * 32;
  const int kb = blockIdx.y * 32;
  const int tx = threadIdx.x, ty = threadIdx.y;
#pragma unroll
  for (int j = 0; j < 4; j++) {
    const int k = kb + ty + j * 8;
    const int n = nb + tx;
    tile[ty + j * 8][tx] = (n < N) ? in[(size_t)k * N + n] : 0.f;
  }
  __syncthreads();
#pragma unroll
  for (int j = 0; j < 4; j++) {
    const int n = nb + ty + j * 8;       // output row (= weight column)
    const int k = kb + tx;
    out[(size_t)n * K + k] = (_Float16)tile[tx][ty + j * 8];
  }
}

// ---------------------------------------------------------------------------
// WMMA GEMM, no LDS / no barriers: C[M,N] = A[M,K](f16) * Wt^T + bias
// (+ residual, optional ReLU).  Wt is f16 column-major [>=gridX*64][K].
// Block = 256 threads = 8 waves; wave computes 32(M) x 64(N); block 256 x 64.
// Per K-step: 4 A b128 + 8 B b128 loads, 8 v_wmma.
// ---------------------------------------------------------------------------
__global__ __launch_bounds__(256)
void gemm_wmma_kernel(const _Float16* __restrict__ A, const _Float16* __restrict__ Bt,
                      const float* __restrict__ bias, const float* __restrict__ resid,
                      float* __restrict__ Cf, _Float16* __restrict__ Ch,
                      int M, int N, int K, int relu)
{
  const int wave = threadIdx.x >> 5;
  const int lane = threadIdx.x & 31;
  const int l    = lane & 15;
  const int hf   = lane >> 4;
  const int nBase = blockIdx.x * 64;
  const int mBase = blockIdx.y * 256;
  const int row0  = mBase + wave * 32 + l;
  const int row1  = row0 + 16;

  v8f acc[2][4];
#pragma unroll
  for (int rt = 0; rt < 2; rt++)
#pragma unroll
    for (int nt = 0; nt < 4; nt++)
#pragma unroll
      for (int e = 0; e < 8; e++) acc[rt][nt][e] = 0.f;

  const _Float16* ap0 = A + (size_t)row0 * K + hf * 8;
  const _Float16* ap1 = A + (size_t)row1 * K + hf * 8;

  for (int k0 = 0; k0 < K; k0 += 32) {
    // A fragments: lane group hf holds K = k0 + hf*8 + {0..7, 16..23}
    const v16h a0 = cat16(ld8(ap0 + k0), ld8(ap0 + k0 + 16));
    const v16h a1 = cat16(ld8(ap1 + k0), ld8(ap1 + k0 + 16));
    __builtin_prefetch(ap0 + k0 + 32, 0, 1);   // global_prefetch_b8 (next K step)
    __builtin_prefetch(ap1 + k0 + 32, 0, 1);
#pragma unroll
    for (int nt = 0; nt < 4; nt++) {
      // B fragment: lane = column, elements = K (hf*8 + {0..7, 16..23})
      const _Float16* bp = Bt + (size_t)(nBase + nt * 16 + l) * K + k0 + hf * 8;
      const v16h bf = cat16(ld8(bp), ld8(bp + 16));
      acc[0][nt] = wmma_f16(a0, bf, acc[0][nt]);
      acc[1][nt] = wmma_f16(a1, bf, acc[1][nt]);
    }
  }

#pragma unroll
  for (int nt = 0; nt < 4; nt++) {
    const int col = nBase + nt * 16 + l;
    if (col >= N) continue;                     // only ragged LM-head edge tile
    const float bv = bias ? bias[col] : 0.f;
#pragma unroll
    for (int rt = 0; rt < 2; rt++)
#pragma unroll
      for (int e = 0; e < 8; e++) {
        const int r = mBase + wave * 32 + rt * 16 + e + 8 * hf;
        float v = acc[rt][nt][e] + bv;
        if (relu) v = fmaxf(v, 0.f);
        if (resid) v += resid[(size_t)r * N + col];
        if (Cf) Cf[(size_t)r * N + col] = v;
        if (Ch) Ch[(size_t)r * N + col] = (_Float16)v;
      }
  }
}

// ---------------------------------------------------------------------------
// Flash attention, one wave per (b, h, 16-row q block). WMMA for Q*K^T and
// P*V, online softmax via 16-lane shfl_xor reductions (C-layout: lane=col,
// element=row). P re-shaped C-layout -> A-layout through LDS.
// ---------------------------------------------------------------------------
__global__ __launch_bounds__(32)
void attn_kernel(const _Float16* __restrict__ q, const _Float16* __restrict__ kmat,
                 const _Float16* __restrict__ vt, _Float16* __restrict__ o)
{
  __shared__ _Float16 ldsP[16 * 32];
  const int D = GPT_D, T = GPT_T, HD = GPT_HD, Hn = GPT_H;
  const int lane = threadIdx.x & 31;
  const int l  = lane & 15;
  const int hf = lane >> 4;
  const int qb = blockIdx.x * 16;
  const int h  = blockIdx.y;
  const int b  = blockIdx.z;
  const float scale = 0.125f;                 // HD^-0.5

  const _Float16* qrow = q + ((size_t)(b * T + qb + l)) * D + h * HD + hf * 8;
  const v16h aq0 = cat16(ld8(qrow),      ld8(qrow + 16));   // K dims 0..31
  const v16h aq1 = cat16(ld8(qrow + 32), ld8(qrow + 48));   // K dims 32..63

  float m[8], lsum[8];
  v8f accO[4];
#pragma unroll
  for (int e = 0; e < 8; e++) { m[e] = -1e30f; lsum[e] = 0.f; }
#pragma unroll
  for (int nt = 0; nt < 4; nt++)
#pragma unroll
    for (int e = 0; e < 8; e++) accO[nt][e] = 0.f;

  for (int sb = 0; sb < qb + 16; sb += 32) {
    v8f s0, s1;
#pragma unroll
    for (int e = 0; e < 8; e++) { s0[e] = 0.f; s1[e] = 0.f; }
    {
      const _Float16* kr = kmat + ((size_t)(b * T + sb + l)) * D + h * HD + hf * 8;
      s0 = wmma_f16(aq0, cat16(ld8(kr),      ld8(kr + 16)), s0);
      s0 = wmma_f16(aq1, cat16(ld8(kr + 32), ld8(kr + 48)), s0);
    }
    {
      const _Float16* kr = kmat + ((size_t)(b * T + sb + 16 + l)) * D + h * HD + hf * 8;
      s1 = wmma_f16(aq0, cat16(ld8(kr),      ld8(kr + 16)), s1);
      s1 = wmma_f16(aq1, cat16(ld8(kr + 32), ld8(kr + 48)), s1);
    }

#pragma unroll
    for (int e = 0; e < 8; e++) {
      const int trow = qb + e + 8 * hf;
      float v0 = s0[e] * scale; if (sb + l      > trow) v0 = -1e30f;
      float v1 = s1[e] * scale; if (sb + 16 + l > trow) v1 = -1e30f;
      float mx = fmaxf(v0, v1);
#pragma unroll
      for (int ofs = 1; ofs < 16; ofs <<= 1) mx = fmaxf(mx, __shfl_xor(mx, ofs, 32));
      const float mn    = fmaxf(m[e], mx);
      const float alpha = __expf(m[e] - mn);
      const float p0 = __expf(v0 - mn);
      const float p1 = __expf(v1 - mn);
      float rs = p0 + p1;
#pragma unroll
      for (int ofs = 1; ofs < 16; ofs <<= 1) rs += __shfl_xor(rs, ofs, 32);
      lsum[e] = lsum[e] * alpha + rs;
      m[e] = mn;
#pragma unroll
      for (int nt = 0; nt < 4; nt++) accO[nt][e] *= alpha;
      const int prow = e + 8 * hf;
      ldsP[prow * 32 + l]      = (_Float16)p0;
      ldsP[prow * 32 + 16 + l] = (_Float16)p1;
    }
    __syncthreads();

    const _Float16* pp = &ldsP[l * 32 + hf * 8];
    const v16h pf = cat16(ld8(pp), ld8(pp + 16));
#pragma unroll
    for (int nt = 0; nt < 4; nt++) {
      const _Float16* vp = vt + (((size_t)(b * Hn + h) * HD) + nt * 16 + l) * T + sb + hf * 8;
      const v16h vf = cat16(ld8(vp), ld8(vp + 16));
      accO[nt] = wmma_f16(pf, vf, accO[nt]);
    }
    __syncthreads();
  }

#pragma unroll
  for (int nt = 0; nt < 4; nt++)
#pragma unroll
    for (int e = 0; e < 8; e++) {
      const int t = qb + e + 8 * hf;
      o[((size_t)(b * T + t)) * D + h * HD + nt * 16 + l] =
          (_Float16)(accO[nt][e] / lsum[e]);
    }
}

// ---------------------------------------------------------------------------
// Elementwise / reduction kernels
// ---------------------------------------------------------------------------
__global__ __launch_bounds__(256)
void embed_kernel(const int* __restrict__ idx, const float* __restrict__ tok,
                  const float* __restrict__ pos, float* __restrict__ x)
{
  const int i = blockIdx.x * 256 + threadIdx.x;
  if (i >= GPT_BT * GPT_D) return;
  const int d = i % GPT_D;
  const int bt = i / GPT_D;
  const int t = bt % GPT_T;
  x[i] = tok[(size_t)idx[bt] * GPT_D + d] + pos[(size_t)t * GPT_D + d];
}

__global__ __launch_bounds__(256)
void ln_f16_kernel(const float* __restrict__ x, const float* __restrict__ g,
                   const float* __restrict__ b, _Float16* __restrict__ out)
{
  __shared__ float red[256];
  const int D = GPT_D;
  const int row = blockIdx.x, tid = threadIdx.x;
  const float* xr = x + (size_t)row * D;
  float s = 0.f;
  for (int i = tid; i < D; i += 256) s += xr[i];
  red[tid] = s; __syncthreads();
  for (int st = 128; st > 0; st >>= 1) { if (tid < st) red[tid] += red[tid + st]; __syncthreads(); }
  const float mean = red[0] * (1.f / D);
  __syncthreads();
  float s2 = 0.f;
  for (int i = tid; i < D; i += 256) { const float d = xr[i] - mean; s2 += d * d; }
  red[tid] = s2; __syncthreads();
  for (int st = 128; st > 0; st >>= 1) { if (tid < st) red[tid] += red[tid + st]; __syncthreads(); }
  const float rstd = rsqrtf(red[0] * (1.f / D) + 1e-5f);
  for (int i = tid; i < D; i += 256)
    out[(size_t)row * D + i] = (_Float16)((xr[i] - mean) * rstd * g[i] + b[i]);
}

// v[b,t,h,hd] (f16) -> vt[b,h,hd,t] (f16); tiled transpose over (t,hd).
// Grid: (T/32, HD/32, B*H), block (32,8).
__global__ __launch_bounds__(256)
void vtrans_kernel(const _Float16* __restrict__ v, _Float16* __restrict__ vt)
{
  __shared__ _Float16 tile[32][33];
  const int tb  = blockIdx.x * 32;
  const int hdb = blockIdx.y * 32;
  const int bh  = blockIdx.z;
  const int b = bh / GPT_H, h = bh % GPT_H;
  const int tx = threadIdx.x, ty = threadIdx.y;
#pragma unroll
  for (int j = 0; j < 4; j++) {
    const int t  = tb + ty + j * 8;
    const int hd = hdb + tx;
    tile[ty + j * 8][tx] = v[((size_t)(b * GPT_T + t)) * GPT_D + h * GPT_HD + hd];
  }
  __syncthreads();
#pragma unroll
  for (int j = 0; j < 4; j++) {
    const int hd = hdb + ty + j * 8;
    const int t  = tb + tx;
    vt[(((size_t)b * GPT_H + h) * GPT_HD + hd) * GPT_T + t] = tile[tx][ty + j * 8];
  }
}

__global__ __launch_bounds__(256)
void cvt_f16_kernel(const float* __restrict__ in, _Float16* __restrict__ out, int n)
{
  const int i = blockIdx.x * 256 + threadIdx.x;
  if (i < n) out[i] = (_Float16)in[i];
}

__global__ void zero_kernel(float* p)
{
  if (blockIdx.x == 0 && threadIdx.x == 0) p[0] = 0.f;
}

__global__ __launch_bounds__(256)
void loss_kernel(const float* __restrict__ logits, const int* __restrict__ targets,
                 float* __restrict__ loss, int rows, int V)
{
  __shared__ float red[256];
  const int row = blockIdx.x, tid = threadIdx.x;
  const float* lr = logits + (size_t)row * V;
  float mx = -3.4e38f;
  for (int i = tid; i < V; i += 256) mx = fmaxf(mx, lr[i]);
  red[tid] = mx; __syncthreads();
  for (int st = 128; st > 0; st >>= 1) { if (tid < st) red[tid] = fmaxf(red[tid], red[tid + st]); __syncthreads(); }
  const float m = red[0];
  __syncthreads();
  float s = 0.f;
  for (int i = tid; i < V; i += 256) s += __expf(lr[i] - m);
  red[tid] = s; __syncthreads();
  for (int st = 128; st > 0; st >>= 1) { if (tid < st) red[tid] += red[tid + st]; __syncthreads(); }
  if (tid == 0) {
    const float lse = m + logf(red[0]);
    const float nll = lse - lr[targets[row]];
    atomicAdd(loss, nll / (float)rows);
  }
}

// ---------------------------------------------------------------------------
// Host launcher
// ---------------------------------------------------------------------------
extern "C" void kernel_launch(void* const* d_in, const int* in_sizes, int n_in,
                              void* d_out, int out_size, void* d_ws, size_t ws_size,
                              hipStream_t stream)
{
  (void)in_sizes; (void)n_in; (void)out_size; (void)ws_size;

  const int*   idx     = (const int*)  d_in[0];
  const int*   targets = (const int*)  d_in[1];
  const float* tok_emb = (const float*)d_in[2];
  const float* pos_emb = (const float*)d_in[3];
  const float* Wq      = (const float*)d_in[4];
  const float* Wk      = (const float*)d_in[5];
  const float* Wv      = (const float*)d_in[6];
  const float* Wo      = (const float*)d_in[7];
  const float* bo      = (const float*)d_in[8];
  const float* ln1_g   = (const float*)d_in[9];
  const float* ln1_b   = (const float*)d_in[10];
  const float* ln2_g   = (const float*)d_in[11];
  const float* ln2_b   = (const float*)d_in[12];
  const float* W1      = (const float*)d_in[13];
  const float* b1      = (const float*)d_in[14];
  const float* W2      = (const float*)d_in[15];
  const float* b2      = (const float*)d_in[16];
  const float* Wlm     = (const float*)d_in[17];
  const float* blm     = (const float*)d_in[18];

  char* ws = (char*)d_ws;
  size_t off = 0;
  auto alloc = [&](size_t bytes) -> void* {
    void* p = ws + off;
    off += (bytes + 255) & ~(size_t)255;
    return p;
  };
  const size_t BT = GPT_BT;
  float*    x    = (float*)   alloc(BT * GPT_D  * sizeof(float));
  float*    x2   = (float*)   alloc(BT * GPT_D  * sizeof(float));
  _Float16* h    = (_Float16*)alloc(BT * GPT_D  * sizeof(_Float16));
  _Float16* qb   = (_Float16*)alloc(BT * GPT_D  * sizeof(_Float16));
  _Float16* kb   = (_Float16*)alloc(BT * GPT_D  * sizeof(_Float16));
  _Float16* vb   = (_Float16*)alloc(BT * GPT_D  * sizeof(_Float16));
  _Float16* vtb  = (_Float16*)alloc(BT * GPT_D  * sizeof(_Float16));
  _Float16* ob   = (_Float16*)alloc(BT * GPT_D  * sizeof(_Float16));
  _Float16* ff1  = (_Float16*)alloc(BT * GPT_DF * sizeof(_Float16));
  // transposed f16 weights (per-layer buffers, reused each layer)
  _Float16* wqt  = (_Float16*)alloc((size_t)GPT_D  * GPT_D  * sizeof(_Float16));
  _Float16* wkt  = (_Float16*)alloc((size_t)GPT_D  * GPT_D  * sizeof(_Float16));
  _Float16* wvt  = (_Float16*)alloc((size_t)GPT_D  * GPT_D  * sizeof(_Float16));
  _Float16* wot  = (_Float16*)alloc((size_t)GPT_D  * GPT_D  * sizeof(_Float16));
  _Float16* w1t  = (_Float16*)alloc((size_t)GPT_DF * GPT_D  * sizeof(_Float16));
  _Float16* w2t  = (_Float16*)alloc((size_t)GPT_D  * GPT_DF * sizeof(_Float16));
  _Float16* wlmt = (_Float16*)alloc((size_t)GPT_VP * GPT_D  * sizeof(_Float16));

  float* logits = (float*)d_out;
  float* lossp  = logits + (size_t)BT * GPT_V;

  auto gemm = [&](const _Float16* A, const _Float16* Bt, const float* bias,
                  const float* resid, float* Cf, _Float16* Ch,
                  int M, int N, int K, int relu) {
    dim3 g((N + 63) / 64, M / 256);
    gemm_wmma_kernel<<<g, 256, 0, stream>>>(A, Bt, bias, resid, Cf, Ch, M, N, K, relu);
  };
  auto wtrans = [&](const float* w, _Float16* wt, int K, int N, int Npad) {
    wtrans_kernel<<<dim3(Npad / 32, K / 32), dim3(32, 8), 0, stream>>>(w, wt, K, N);
  };

  const int ED = GPT_BT * GPT_D;
  embed_kernel<<<(ED + 255) / 256, 256, 0, stream>>>(idx, tok_emb, pos_emb, x);

  for (int l = 0; l < GPT_L; l++) {
    const float* wq = Wq + (size_t)l * GPT_D * GPT_D;
    const float* wk = Wk + (size_t)l * GPT_D * GPT_D;
    const float* wv = Wv + (size_t)l * GPT_D * GPT_D;
    const float* wo = Wo + (size_t)l * GPT_D * GPT_D;
    const float* w1 = W1 + (size_t)l * GPT_D * GPT_DF;
    const float* w2 = W2 + (size_t)l * GPT_DF * GPT_D;

    wtrans(wq, wqt, GPT_D,  GPT_D,  GPT_D);
    wtrans(wk, wkt, GPT_D,  GPT_D,  GPT_D);
    wtrans(wv, wvt, GPT_D,  GPT_D,  GPT_D);
    wtrans(wo, wot, GPT_D,  GPT_D,  GPT_D);
    wtrans(w1, w1t, GPT_D,  GPT_DF, GPT_DF);
    wtrans(w2, w2t, GPT_DF, GPT_D,  GPT_D);

    ln_f16_kernel<<<GPT_BT, 256, 0, stream>>>(x, ln1_g + l * GPT_D, ln1_b + l * GPT_D, h);
    gemm(h, wqt, nullptr, nullptr, nullptr, qb, GPT_BT, GPT_D, GPT_D, 0);
    gemm(h, wkt, nullptr, nullptr, nullptr, kb, GPT_BT, GPT_D, GPT_D, 0);
    gemm(h, wvt, nullptr, nullptr, nullptr, vb, GPT_BT, GPT_D, GPT_D, 0);
    vtrans_kernel<<<dim3(GPT_T / 32, GPT_HD / 32, GPT_B * GPT_H), dim3(32, 8), 0, stream>>>(vb, vtb);
    attn_kernel<<<dim3(GPT_T / 16, GPT_H, GPT_B), 32, 0, stream>>>(qb, kb, vtb, ob);
    gemm(ob, wot, bo + l * GPT_D, x, x2, nullptr, GPT_BT, GPT_D, GPT_D, 0);
    ln_f16_kernel<<<GPT_BT, 256, 0, stream>>>(x2, ln2_g + l * GPT_D, ln2_b + l * GPT_D, h);
    gemm(h, w1t, b1 + (size_t)l * GPT_DF, nullptr, nullptr, ff1, GPT_BT, GPT_DF, GPT_D, 1);
    gemm(ff1, w2t, b2 + l * GPT_D, x2, x, nullptr, GPT_BT, GPT_D, GPT_DF, 0);
  }

  // LM head (no final LN in reference) + loss
  wtrans(Wlm, wlmt, GPT_D, GPT_V, GPT_VP);
  cvt_f16_kernel<<<(ED + 255) / 256, 256, 0, stream>>>(x, h, ED);
  gemm(h, wlmt, blm, nullptr, logits, nullptr, GPT_BT, GPT_V, GPT_D, 0);
  zero_kernel<<<1, 32, 0, stream>>>(lossp);
  loss_kernel<<<GPT_BT, 256, 0, stream>>>(logits, targets, lossp, GPT_BT, GPT_V);
}
